// PcConvBp_SGD_14078902796548
// MI455X (gfx1250) — matrix operand order repaired and checked by reading, not compiled, not run on hardware
//
#include <hip/hip_runtime.h>

// ---------------------------------------------------------------------------
// Problem constants (from reference): B=4, CIN=64, COUT=128, H=W=32, K=3, P=1
// ---------------------------------------------------------------------------
#define NUM_ITERS 500
#define LR_F      0.001f

static const int NB    = 4;
static const int NPIXP = NB * 34 * 34;            // 4624  (bwd GEMM M)
static const int NY    = NB * 128 * 34 * 34;      // 591872 yp elements
static const int NR    = NB * 64 * 32 * 32;       // 262144 residual elements
static const int NOUT  = NB * 128 * 32 * 32;      // 524288 output elements
static const int NPART = 128;                     // fwd grid = #partials

typedef __attribute__((ext_vector_type(16))) __bf16 v16bf;
typedef __attribute__((ext_vector_type(8)))  __bf16 v8bf;
typedef __attribute__((ext_vector_type(8)))  float  v8f;
typedef __attribute__((ext_vector_type(4)))  int    v4i;

// pointer-to-AS1-int4 and pointer-to-AS3-int4, as the async builtin expects
typedef __attribute__((address_space(1))) v4i* gptr_v4i;
typedef __attribute__((address_space(3))) v4i* lptr_v4i;

union V16 { v16bf v; v8bf h[2]; };

// ---------------------------------------------------------------------------
// CDNA5 async global->LDS copy (ASYNCcnt-tracked), with VGPR fallback.
// ---------------------------------------------------------------------------
#if __has_builtin(__builtin_amdgcn_global_load_async_to_lds_b128) && \
    __has_builtin(__builtin_amdgcn_s_wait_asynccnt)
#define HAVE_ASYNC 1
#else
#define HAVE_ASYNC 0
#endif

__device__ __forceinline__ void async_b128(const void* gsrc, void* lds) {
#if HAVE_ASYNC
    __builtin_amdgcn_global_load_async_to_lds_b128(
        (gptr_v4i)(unsigned long long)gsrc,   // int-cast sheds const, AS1 ptr
        (lptr_v4i)lds, 0, 0);
#else
    *(v8bf*)lds = *(const v8bf*)gsrc;
#endif
}
__device__ __forceinline__ void async_wait0() {
#if HAVE_ASYNC
    __builtin_amdgcn_s_wait_asynccnt(0);
#endif
}

// ---------------------------------------------------------------------------
// Repack W_fb (f32, [ci=128][co=64][3][3]) into bf16 GEMM-ready layouts:
//   Wfwd[(kk*4+cb)*64 + co][cil]   = W_fb[ci=cb*32+cil, co, kh, kw]
//   Wbwd[(kk*2+cob)*128 + ci][col] = W_fb[ci, co=cob*32+col, 2-k1, 2-k2]
// ---------------------------------------------------------------------------
__global__ __launch_bounds__(256) void k_repack(const float* __restrict__ Wfb,
                                                __bf16* __restrict__ Wfwd,
                                                __bf16* __restrict__ Wbwd) {
    int t = blockIdx.x * 256 + threadIdx.x;
    if (t >= 9 * 4 * 64 * 32) return;   // 73728 == 9*2*128*32 as well
    {
        int cil = t & 31;
        int co  = (t >> 5) & 63;
        int rest = t >> 11;             // kk*4+cb, < 36
        int cb = rest & 3, kk = rest >> 2;
        int kh = kk / 3, kw = kk % 3;
        int ci = cb * 32 + cil;
        Wfwd[t] = (__bf16)Wfb[((ci * 64 + co) * 3 + kh) * 3 + kw];
    }
    {
        int col = t & 31;
        int ci  = (t >> 5) & 127;
        int rest = t >> 12;             // kk*2+cob, < 18
        int cob = rest & 1, kk = rest >> 1;
        int k1 = kk / 3, k2 = kk % 3;
        int co = cob * 32 + col;
        Wbwd[t] = (__bf16)Wfb[((ci * 64 + co) * 3 + (2 - k1)) * 3 + (2 - k2)];
    }
}

// ---------------------------------------------------------------------------
// Init: yp = pad( relu( conv(x, W_ff, pad=1) ) ), border ring = 0.  One-shot.
// ---------------------------------------------------------------------------
__global__ __launch_bounds__(256) void k_init(const float* __restrict__ x,
                                              const float* __restrict__ Wff,
                                              float* __restrict__ yp) {
    int idx = blockIdx.x * 256 + threadIdx.x;
    if (idx >= NY) return;
    int xx = idx % 34;
    int tt = idx / 34;
    int yy = tt % 34; tt /= 34;
    int ci = tt % 128;
    int b  = tt / 128;
    float val = 0.f;
    if (xx >= 1 && xx < 33 && yy >= 1 && yy < 33) {
        int h = yy - 1, w = xx - 1;
        float s = 0.f;
        for (int c = 0; c < 64; ++c) {
            const float* xc = x + ((size_t)(b * 64 + c) * 32) * 32;
            const float* wc = Wff + ((size_t)(ci * 64 + c) * 9);
            #pragma unroll
            for (int kh = 0; kh < 3; ++kh) {
                int ih = h - 1 + kh;
                if ((unsigned)ih >= 32u) continue;
                #pragma unroll
                for (int kw = 0; kw < 3; ++kw) {
                    int iw = w - 1 + kw;
                    if ((unsigned)iw >= 32u) continue;
                    s += xc[ih * 32 + iw] * wc[kh * 3 + kw];
                }
            }
        }
        val = fmaxf(s, 0.f);
    }
    yp[idx] = val;
}

// ---------------------------------------------------------------------------
// Forward conv via WMMA:  r = x - conv(yp, Wk), partials[blk] = local sum r^2.
// Grid: 128 blocks = (b,h); 8 waves = 2 Mtiles x 4 Ntiles.
// GEMM M=32 (w), N=64 (co), K=1152 in 36 steps of 32; double-buffered LDS,
// software-pipelined A (regs) and B (async global->LDS), 1 barrier/step.
// ---------------------------------------------------------------------------
__global__ __launch_bounds__(256) void k_fwd(const float* __restrict__ yp,
                                             const __bf16* __restrict__ Wfwd,
                                             const float* __restrict__ x,
                                             float* __restrict__ r,
                                             float* __restrict__ partials) {
    __shared__ __bf16 lA[2][32][32];     // [buf][m=w'][k=ci_local]
    __shared__ __bf16 lB[2][64 * 32];    // [buf][n=co][k=ci_local]
    __shared__ float  sred[256];

    const int t    = threadIdx.x;
    const int b    = blockIdx.x >> 5;
    const int h    = blockIdx.x & 31;
    const int lane = t & 31, lo = lane & 15, hi = lane >> 4;
    const int wave = t >> 5;
    const int mtile = wave & 1, ntile = wave >> 1;

    const int cil = t >> 3;          // 0..31 : ci_local this thread loads
    const int wq  = (t & 7) * 4;     // 0..28 : 4 consecutive w' values

    // A source address for K-step s: kk = s>>2 (kh,kw), cb = s&3 (ci block)
    auto aSrc = [&](int s) -> const float* {
        const int kk = s >> 2, cb = s & 3;
        const int kh = kk / 3, kw = kk % 3;
        const int ci = cb * 32 + cil;
        return yp + (((size_t)(b * 128 + ci) * 34 + (h + kh)) * 34) + kw + wq;
    };

    // ---- prologue: stage step 0
    float a0, a1, a2, a3;
    { const float* sp = aSrc(0); a0 = sp[0]; a1 = sp[1]; a2 = sp[2]; a3 = sp[3]; }
    async_b128(Wfwd + t * 8, &lB[0][t * 8]);                    // B(0)

    v8f acc = {};
    for (int s = 0; s < 36; ++s) {
        const int buf = s & 1;
        // stage A(s) regs -> LDS (bf16, transposed)
        lA[buf][wq + 0][cil] = (__bf16)a0;
        lA[buf][wq + 1][cil] = (__bf16)a1;
        lA[buf][wq + 2][cil] = (__bf16)a2;
        lA[buf][wq + 3][cil] = (__bf16)a3;
        // prefetch A(s+1) into regs (consumed next iteration)
        if (s + 1 < 36) {
            const float* sp = aSrc(s + 1);
            a0 = sp[0]; a1 = sp[1]; a2 = sp[2]; a3 = sp[3];
        }
        async_wait0();                 // B(s) resident in lB[buf]
        __syncthreads();
        // safe to overwrite the other buffer now: everyone is past step s-1
        if (s + 1 < 36)
            async_b128(Wfwd + (size_t)(s + 1) * 2048 + t * 8, &lB[buf ^ 1][t * 8]);

        // WMMA operands in contiguous k order, k = 16*hi + s for A and B (2x ds_load_b128 per half)
        V16 a, bb;
        const __bf16* Ar = &lA[buf][mtile * 16 + lo][0];
        a.h[0] = *(const v8bf*)&Ar[hi * 16];
        a.h[1] = *(const v8bf*)&Ar[hi * 16 + 8];
        const __bf16* Br = &lB[buf][(ntile * 16 + lo) * 32];
        bb.h[0] = *(const v8bf*)&Br[hi * 16];
        bb.h[1] = *(const v8bf*)&Br[hi * 16 + 8];
        acc = __builtin_amdgcn_wmma_f32_16x16x32_bf16(
            false, a.v, false, bb.v, (short)0, acc, false, false);
    }

    // r = x - pred  (C layout: lane -> n = lo, vgpr v -> m = v + 8*hi)
    const int co = ntile * 16 + lo;
    float ss = 0.f;
    #pragma unroll
    for (int v = 0; v < 8; ++v) {
        const int w = mtile * 16 + hi * 8 + v;
        const size_t idx = (((size_t)(b * 64 + co)) * 32 + h) * 32 + w;
        const float rv = x[idx] - acc[v];
        r[idx] = rv;
        ss += rv * rv;
    }
    // deterministic fixed-pairing tree reduction
    sred[t] = ss;
    __syncthreads();
    #pragma unroll
    for (int o = 128; o > 0; o >>= 1) {
        if (t < o) sred[t] += sred[t + o];
        __syncthreads();
    }
    if (t == 0) partials[blockIdx.x] = sred[0];
}

// ---------------------------------------------------------------------------
// Backward conv via WMMA + SGD update:
//   g[b,ci,y,x] = sum_{co,k1,k2} r[b,co,y-2+k1,x-2+k2] * Wfb[ci,co,2-k1,2-k2]
//   yp += (LR / sqrt(sum r^2)) * g
// Grid: 145 blocks of 32 flat padded pixels; 8 waves = 2 Mtiles x 4 Npairs.
// GEMM M=32, N=128 (2 accumulators/wave), K=576 in 18 steps of 32; same
// double-buffered pipeline as k_fwd.
// ---------------------------------------------------------------------------
__global__ __launch_bounds__(256) void k_bwd(const float* __restrict__ r,
                                             const __bf16* __restrict__ Wbwd,
                                             const float* __restrict__ partials,
                                             float* __restrict__ yp) {
    __shared__ __bf16 lA[2][32][32];     // [buf][m_local][k=co_local]
    __shared__ __bf16 lB[2][128 * 32];   // [buf][n=ci][k=co_local]
    __shared__ float  sScale;

    const int t = threadIdx.x;
    if (t == 0) {                        // deterministic: e = sqrt(sum partials)
        float s = 0.f;
        for (int i = 0; i < NPART; ++i) s += partials[i];
        sScale = LR_F / sqrtf(s);
    }

    const int col_  = t >> 3;            // co_local 0..31
    const int mq    = (t & 7) * 4;       // 4 consecutive m rows
    const int mbase = blockIdx.x * 32;

    int pb[4], py[4], px[4]; bool pv[4];
    #pragma unroll
    for (int i = 0; i < 4; ++i) {
        int m = mbase + mq + i;
        pv[i] = (m < NPIXP);
        int bq = m / 1156;
        int rem = m - bq * 1156;
        py[i] = rem / 34;
        px[i] = rem - py[i] * 34;
        pb[i] = bq;
    }

    const int lane = t & 31, lo = lane & 15, hi = lane >> 4;
    const int wave = t >> 5;
    const int mtile = wave & 1, npair = wave >> 1;

    // zero-padded residual gather for K-step s: kk = s>>1, cob = s&1
    auto loadA = [&](int s, float* v) {
        const int kk = s >> 1, cob = s & 1;
        const int k1 = kk / 3, k2 = kk % 3;
        #pragma unroll
        for (int i = 0; i < 4; ++i) {
            const int u  = py[i] - 2 + k1;
            const int vv = px[i] - 2 + k2;
            float val = 0.f;
            if (pv[i] && (unsigned)u < 32u && (unsigned)vv < 32u)
                val = r[(((size_t)(pb[i] * 64 + cob * 32 + col_)) * 32 + u) * 32 + vv];
            v[i] = val;
        }
    };

    // ---- prologue: stage step 0
    float av[4];
    loadA(0, av);
    async_b128(Wbwd + t * 16,     &lB[0][t * 16]);
    async_b128(Wbwd + t * 16 + 8, &lB[0][t * 16 + 8]);

    v8f acc0 = {}, acc1 = {};
    for (int s = 0; s < 18; ++s) {
        const int buf = s & 1;
        lA[buf][mq + 0][col_] = (__bf16)av[0];
        lA[buf][mq + 1][col_] = (__bf16)av[1];
        lA[buf][mq + 2][col_] = (__bf16)av[2];
        lA[buf][mq + 3][col_] = (__bf16)av[3];
        if (s + 1 < 18) loadA(s + 1, av);
        async_wait0();                 // B(s) resident in lB[buf]
        __syncthreads();
        if (s + 1 < 18) {
            const __bf16* src = Wbwd + (size_t)(s + 1) * 4096;
            async_b128(src + t * 16,     &lB[buf ^ 1][t * 16]);
            async_b128(src + t * 16 + 8, &lB[buf ^ 1][t * 16 + 8]);
        }
        V16 a, b0, b1;
        const __bf16* Ar = &lA[buf][mtile * 16 + lo][0];
        a.h[0] = *(const v8bf*)&Ar[hi * 16];
        a.h[1] = *(const v8bf*)&Ar[hi * 16 + 8];
        const __bf16* Br0 = &lB[buf][(npair * 32 + lo) * 32];
        b0.h[0] = *(const v8bf*)&Br0[hi * 16];
        b0.h[1] = *(const v8bf*)&Br0[hi * 16 + 8];
        const __bf16* Br1 = &lB[buf][(npair * 32 + 16 + lo) * 32];
        b1.h[0] = *(const v8bf*)&Br1[hi * 16];
        b1.h[1] = *(const v8bf*)&Br1[hi * 16 + 8];
        acc0 = __builtin_amdgcn_wmma_f32_16x16x32_bf16(
            false, a.v, false, b0.v, (short)0, acc0, false, false);
        acc1 = __builtin_amdgcn_wmma_f32_16x16x32_bf16(
            false, a.v, false, b1.v, (short)0, acc1, false, false);
    }

    const float scale = sScale;          // t0 wrote before first barrier
    const int ci0 = (npair * 2) * 16 + lo;
    const int ci1 = ci0 + 16;
    #pragma unroll
    for (int v = 0; v < 8; ++v) {
        const int m = mbase + mtile * 16 + hi * 8 + v;
        if (m < NPIXP) {
            int bq = m / 1156;
            int rem = m - bq * 1156;
            int y = rem / 34;
            int xx = rem - y * 34;
            size_t i0 = (((size_t)(bq * 128 + ci0)) * 34 + y) * 34 + xx;
            size_t i1 = (((size_t)(bq * 128 + ci1)) * 34 + y) * 34 + xx;
            yp[i0] += scale * acc0[v];
            yp[i1] += scale * acc1[v];
        }
    }
}

// ---------------------------------------------------------------------------
// Final: out = yp interior + 1x1 bypass conv(x, W_bypass)
// ---------------------------------------------------------------------------
__global__ __launch_bounds__(256) void k_final(const float* __restrict__ yp,
                                               const float* __restrict__ x,
                                               const float* __restrict__ Wbp,
                                               float* __restrict__ out) {
    int idx = blockIdx.x * 256 + threadIdx.x;
    if (idx >= NOUT) return;
    int w  = idx % 32;
    int tt = idx / 32;
    int h  = tt % 32; tt /= 32;
    int co = tt % 128;
    int b  = tt / 128;
    float s = yp[(((size_t)(b * 128 + co)) * 34 + (h + 1)) * 34 + (w + 1)];
    const float* xb = x + (((size_t)b * 64) * 32 + h) * 32 + w;
    const float* wb = Wbp + co * 64;
    #pragma unroll 4
    for (int c = 0; c < 64; ++c)
        s += xb[(size_t)c * 1024] * wb[c];
    out[idx] = s;
}

// ---------------------------------------------------------------------------
extern "C" void kernel_launch(void* const* d_in, const int* in_sizes, int n_in,
                              void* d_out, int out_size, void* d_ws, size_t ws_size,
                              hipStream_t stream) {
    const float* x   = (const float*)d_in[0];
    const float* Wff = (const float*)d_in[1];
    const float* Wfb = (const float*)d_in[2];
    const float* Wbp = (const float*)d_in[3];
    float* out = (float*)d_out;

    // workspace carve-up (all 16B aligned)
    float*  yp       = (float*)d_ws;                 // NY f32
    float*  r        = yp + NY;                      // NR f32
    float*  partials = r + NR;                       // NPART f32
    __bf16* Wfwd     = (__bf16*)(partials + NPART);  // 73728 bf16
    __bf16* Wbwd     = Wfwd + 73728;                 // 73728 bf16

    k_repack<<<(73728 + 255) / 256, 256, 0, stream>>>(Wfb, Wfwd, Wbwd);
    k_init<<<(NY + 255) / 256, 256, 0, stream>>>(x, Wff, yp);

    const int bwdGrid = (NPIXP + 31) / 32;           // 145
    for (int it = 0; it < NUM_ITERS; ++it) {
        k_fwd<<<NPART, 256, 0, stream>>>(yp, Wfwd, x, r, partials);
        k_bwd<<<bwdGrid, 256, 0, stream>>>(r, Wbwd, partials, yp);
    }

    k_final<<<(NOUT + 255) / 256, 256, 0, stream>>>(yp, x, Wbp, out);
}
